// Predictor_54537494724716
// MI455X (gfx1250) — matrix operand order repaired and checked by reading, not compiled, not run on hardware
//
#include <hip/hip_runtime.h>

// ---------------------------------------------------------------------------
// MI455X (gfx1250) implementation.
//  - All large GEMMs run on v_wmma_f32_16x16x32_bf16 (fp32 accumulate).
//  - Weights transposed+converted to bf16 [N][K] once per launch.
//  - Heavy GEMMs (M,N multiples of 128): GLOBAL_LOAD_ASYNC_TO_LDS_B128
//    pipeline with 4 LDS buffers / 3 tiles in flight (in-order ASYNCcnt
//    completion lets us wait "<= 8" to retire only the oldest tile).
//  - Ragged GEMMs (M=32 rows, N=200/10000): guarded synchronous staging.
//  - Fused epilogues: bias+GELU->bf16 (hidden), bias->f32 (outputs, with
//    batched output mapping straight into prefA/prefB), bias+residual (FFN).
// ---------------------------------------------------------------------------

typedef __bf16 bf16_t;
typedef __attribute__((ext_vector_type(16))) __bf16 v16bf;
typedef __attribute__((ext_vector_type(8)))  float  v8f;

union FragBF {
    uint4 u[2];
    v16bf v;
};

__device__ __forceinline__ bf16_t f32_to_bf16_rne(float f) {
    unsigned u = __float_as_uint(f);
    unsigned r = (u + 0x7FFFu + ((u >> 16) & 1u)) >> 16;
    unsigned short h = (unsigned short)r;
    bf16_t out;
    __builtin_memcpy(&out, &h, 2);
    return out;
}

__device__ __forceinline__ float gelu_exact(float x) {
    return 0.5f * x * (1.0f + erff(x * 0.70710678118654752440f));
}

// Low 32 bits of a generic pointer to __shared__ data == LDS byte offset
// (flat aperture rule: the LDS address is the low 32 address bits).
__device__ __forceinline__ unsigned lds_off(const void* p) {
    return (unsigned)(unsigned long long)p;
}

// GLOBAL_LOAD_ASYNC_TO_LDS_B128: copies 16B memory->LDS per lane, ASYNCcnt.
__device__ __forceinline__ void async_copy_b128(unsigned lds_addr,
                                                const void* gaddr) {
    unsigned long long g = (unsigned long long)gaddr;
    asm volatile("global_load_async_to_lds_b128 %0, %1, off"
                 :: "v"(lds_addr), "v"(g) : "memory");
}

__device__ __forceinline__ void wait_asynccnt0() {
    asm volatile("s_wait_asynccnt 0x0" ::: "memory");
}
// ASYNC loads complete in order: waiting ASYNCcnt<=8 retires everything but
// the most recent 8 copies (= the 2 newest 4-copy tiles still streaming).
__device__ __forceinline__ void wait_asynccnt8() {
    asm volatile("s_wait_asynccnt 0x8" ::: "memory");
}

// ---------------------------------------------------------------------------
// Shared epilogue.
// Output index = (r/rows_per_batch)*c_batch_stride + (r%rows_per_batch)*N + n
//   EP = 0: f32 out, bias
//   EP = 1: f32 out, bias + residual (residual uses plain r*N+n indexing)
//   EP = 2: bf16 out, bias + exact GELU
// ---------------------------------------------------------------------------
template <int EP>
__device__ __forceinline__ void gemm_epilogue(
    v8f (&acc)[2][4], const float* bias, const float* resid, void* Cout,
    int M, int N, int rows_per_batch, long long c_batch_stride,
    int blockM, int blockN, int wm, int wn, int lane)
{
    const int colL  = lane & 15;
    const int rowHi = (lane >> 4) * 8;
#pragma unroll
    for (int mt = 0; mt < 2; ++mt) {
#pragma unroll
        for (int nt = 0; nt < 4; ++nt) {
            const int col = blockN + wn * 64 + nt * 16 + colL;
            if (col >= N) continue;
            const float bz = bias ? bias[col] : 0.0f;
#pragma unroll
            for (int j = 0; j < 8; ++j) {
                const int r = blockM + wm * 32 + mt * 16 + rowHi + j;
                if (r >= M) continue;
                float v = acc[mt][nt][j] + bz;
                const size_t idx =
                    (size_t)(r / rows_per_batch) * (size_t)c_batch_stride +
                    (size_t)(r % rows_per_batch) * (size_t)N + (size_t)col;
                if (EP == 0) {
                    ((float*)Cout)[idx] = v;
                } else if (EP == 1) {
                    ((float*)Cout)[idx] = v + resid[(size_t)r * N + col];
                } else {
                    ((bf16_t*)Cout)[idx] = f32_to_bf16_rne(gelu_exact(v));
                }
            }
        }
    }
}

// Fragment loads + 8 WMMAs for one 32-deep K step of one wave tile.
__device__ __forceinline__ void wmma_step(
    const bf16_t* Asl, const bf16_t* Bsl, v8f (&acc)[2][4],
    int wm, int wn, int lane, int LDT)
{
    // A 16x32 bf16: lanes 0-15 hold K{0-7,16-23}; lanes 16-31 K{8-15,24-31}
    const int kha = (lane >> 4) * 8;
    FragBF afrag[2];
#pragma unroll
    for (int mt = 0; mt < 2; ++mt) {
        const int r = wm * 32 + mt * 16 + (lane & 15);
        afrag[mt].u[0] = *(const uint4*)(&Asl[r * LDT + kha]);
        afrag[mt].u[1] = *(const uint4*)(&Asl[r * LDT + kha + 16]);
    }
    // B 32x16 bf16: lanes 0-15 hold K{0-7,16-23}; lanes 16-31 K{8-15,24-31}
    const int khb = (lane >> 4) * 8;
#pragma unroll
    for (int nt = 0; nt < 4; ++nt) {
        FragBF bfrag;
        const int n = wn * 64 + nt * 16 + (lane & 15);
        bfrag.u[0] = *(const uint4*)(&Bsl[n * LDT + khb]);
        bfrag.u[1] = *(const uint4*)(&Bsl[n * LDT + khb + 16]);
#pragma unroll
        for (int mt = 0; mt < 2; ++mt) {
            acc[mt][nt] = __builtin_amdgcn_wmma_f32_16x16x32_bf16(
                false, afrag[mt].v, false, bfrag.v,
                (short)0, acc[mt][nt], false, false);
        }
    }
}

// ---------------------------------------------------------------------------
// ASYNC variant: requires M%128==0, N%128==0, K%32==0 (no guards).
// 4 LDS buffers, 3 tiles in flight via async-to-LDS.
// ---------------------------------------------------------------------------
template <int EP>
__global__ __launch_bounds__(256) void gemm_bf16_wmma_async(
    const bf16_t* __restrict__ A, const bf16_t* __restrict__ Bt,
    const float* __restrict__ bias, const float* __restrict__ resid,
    void* __restrict__ Cout,
    int M, int N, int K, int rows_per_batch, long long c_batch_stride)
{
    constexpr int BM = 128, BN = 128, BK = 32, LDT = 40;
    constexpr int NBUF = 4, IN_FLIGHT = 3;
    __shared__ bf16_t Asl[NBUF][BM * LDT];
    __shared__ bf16_t Bsl[NBUF][BM * LDT];

    const int tid  = threadIdx.x;
    const int lane = tid & 31;
    const int wave = tid >> 5;
    const int wm = wave >> 1;
    const int wn = wave & 1;
    const int blockM = blockIdx.y * BM;
    const int blockN = blockIdx.x * BN;

    // staging coordinates: 2 chunks of 8 bf16 (16B) per matrix per thread
    const int r0 = tid >> 2;              // rows 0..63
    const int kc0 = (tid & 3) * 8;        // 0,8,16,24
    const int r1 = r0 + 64;               // rows 64..127

    v8f acc[2][4];
#pragma unroll
    for (int i = 0; i < 2; ++i)
#pragma unroll
        for (int j = 0; j < 4; ++j)
#pragma unroll
            for (int e = 0; e < 8; ++e) acc[i][j][e] = 0.0f;

    auto issue_tile = [&](int kt, int bufi) {
        const size_t k0 = (size_t)kt * BK + kc0;
        async_copy_b128(lds_off(&Asl[bufi][r0 * LDT + kc0]),
                        A + (size_t)(blockM + r0) * K + k0);
        async_copy_b128(lds_off(&Asl[bufi][r1 * LDT + kc0]),
                        A + (size_t)(blockM + r1) * K + k0);
        async_copy_b128(lds_off(&Bsl[bufi][r0 * LDT + kc0]),
                        Bt + (size_t)(blockN + r0) * K + k0);
        async_copy_b128(lds_off(&Bsl[bufi][r1 * LDT + kc0]),
                        Bt + (size_t)(blockN + r1) * K + k0);
    };

    const int nK = K / BK;
#pragma unroll
    for (int p = 0; p < IN_FLIGHT; ++p)
        if (p < nK) issue_tile(p, p & (NBUF - 1));

    for (int kt = 0; kt < nK; ++kt) {
        const int buf = kt & (NBUF - 1);
        // Retire tile kt: if tiles kt+1 and kt+2 are both still streaming
        // (8 newer copies), ASYNCcnt<=8 is exactly "tile kt landed".
        if (kt + 2 < nK) wait_asynccnt8();
        else             wait_asynccnt0();
        __syncthreads();   // all waves' copies into buf visible; all waves
                           // done reading the buffer we are about to refill
        if (kt + IN_FLIGHT < nK)
            issue_tile(kt + IN_FLIGHT, (kt + IN_FLIGHT) & (NBUF - 1));
        wmma_step(&Asl[buf][0], &Bsl[buf][0], acc, wm, wn, lane, LDT);
    }

    gemm_epilogue<EP>(acc, bias, resid, Cout, M, N, rows_per_batch,
                      c_batch_stride, blockM, blockN, wm, wn, lane);
}

// ---------------------------------------------------------------------------
// Guarded variant for ragged shapes (M=32 rows, N=200/10000).
// ---------------------------------------------------------------------------
template <int EP>
__global__ __launch_bounds__(256) void gemm_bf16_wmma_guard(
    const bf16_t* __restrict__ A, const bf16_t* __restrict__ Bt,
    const float* __restrict__ bias, const float* __restrict__ resid,
    void* __restrict__ Cout,
    int M, int N, int K, int rows_per_batch, long long c_batch_stride)
{
    constexpr int BM = 128, BN = 128, BK = 32, LDT = 40;
    __shared__ bf16_t Asl[BM * LDT];
    __shared__ bf16_t Bsl[BM * LDT];

    const int tid  = threadIdx.x;
    const int lane = tid & 31;
    const int wave = tid >> 5;
    const int wm = wave >> 1;
    const int wn = wave & 1;
    const int blockM = blockIdx.y * BM;
    const int blockN = blockIdx.x * BN;

    v8f acc[2][4];
#pragma unroll
    for (int i = 0; i < 2; ++i)
#pragma unroll
        for (int j = 0; j < 4; ++j)
#pragma unroll
            for (int e = 0; e < 8; ++e) acc[i][j][e] = 0.0f;

    const int nK = K / BK;
    for (int kt = 0; kt < nK; ++kt) {
        const int k0 = kt * BK;
#pragma unroll
        for (int it = 0; it < 2; ++it) {
            const int c   = tid + it * 256;
            const int row = c >> 2;
            const int kc  = (c & 3) * 8;
            uint4 va{0u, 0u, 0u, 0u};
            const int gr = blockM + row;
            if (gr < M) va = *(const uint4*)(A + (size_t)gr * K + (k0 + kc));
            *(uint4*)(&Asl[row * LDT + kc]) = va;

            uint4 vb{0u, 0u, 0u, 0u};
            const int gn = blockN + row;
            if (gn < N) vb = *(const uint4*)(Bt + (size_t)gn * K + (k0 + kc));
            *(uint4*)(&Bsl[row * LDT + kc]) = vb;
        }
        __syncthreads();
        wmma_step(Asl, Bsl, acc, wm, wn, lane, LDT);
        __syncthreads();
    }

    gemm_epilogue<EP>(acc, bias, resid, Cout, M, N, rows_per_batch,
                      c_batch_stride, blockM, blockN, wm, wn, lane);
}

// ---------------------------------------------------------------------------
// Transpose + downconvert: in f32 [R][C] -> out bf16 [C][R]
// ---------------------------------------------------------------------------
__global__ __launch_bounds__(256) void transpose_cvt(
    const float* __restrict__ in, bf16_t* __restrict__ out, int R, int C)
{
    __shared__ float tile[32][33];
    const int tc = blockIdx.x * 32;
    const int tr = blockIdx.y * 32;
    const int tx = threadIdx.x & 31;
    const int ty = threadIdx.x >> 5;
#pragma unroll
    for (int j = 0; j < 32; j += 8) {
        const int r = tr + ty + j, c = tc + tx;
        if (r < R && c < C) tile[ty + j][tx] = in[(size_t)r * C + c];
    }
    __syncthreads();
#pragma unroll
    for (int j = 0; j < 32; j += 8) {
        const int r = tc + ty + j;
        const int c = tr + tx;
        if (r < C && c < R) out[(size_t)r * R + c] = f32_to_bf16_rne(tile[tx][ty + j]);
    }
}

__global__ __launch_bounds__(256) void cvt_f32_bf16(
    const float* __restrict__ in, bf16_t* __restrict__ out, int n)
{
    const int i = blockIdx.x * 256 + threadIdx.x;
    if (i < n) out[i] = f32_to_bf16_rne(in[i]);
}

// ---------------------------------------------------------------------------
// Row LayerNorm (biased var) -> bf16 output. One block (256 thr) per row.
// ---------------------------------------------------------------------------
__global__ __launch_bounds__(256) void layernorm_to_bf16(
    const float* __restrict__ x, const float* __restrict__ g,
    const float* __restrict__ b, bf16_t* __restrict__ y, int H)
{
    __shared__ float red[256];
    const int row = blockIdx.x;
    const float* xr = x + (size_t)row * H;
    float s = 0.0f, s2 = 0.0f;
    for (int i = threadIdx.x; i < H; i += 256) { float v = xr[i]; s += v; s2 += v * v; }
    red[threadIdx.x] = s; __syncthreads();
    for (int o = 128; o; o >>= 1) { if (threadIdx.x < o) red[threadIdx.x] += red[threadIdx.x + o]; __syncthreads(); }
    const float mean = red[0] / H; __syncthreads();
    red[threadIdx.x] = s2; __syncthreads();
    for (int o = 128; o; o >>= 1) { if (threadIdx.x < o) red[threadIdx.x] += red[threadIdx.x + o]; __syncthreads(); }
    const float var = red[0] / H - mean * mean;
    const float rs = rsqrtf(var + 1e-5f);
    for (int i = threadIdx.x; i < H; i += 256) {
        float v = (xr[i] - mean) * rs * g[i] + b[i];
        y[(size_t)row * H + i] = f32_to_bf16_rne(v);
    }
}

// ---------------------------------------------------------------------------
// Final traffic head: per row, LN(ln2) then dot with w[H] + bias -> out[row]
// ---------------------------------------------------------------------------
__global__ __launch_bounds__(256) void traffic_head(
    const float* __restrict__ t, const float* __restrict__ g,
    const float* __restrict__ b, const float* __restrict__ w,
    const float* __restrict__ bias, float* __restrict__ outp, int H)
{
    __shared__ float red[256];
    const int row = blockIdx.x;
    const float* xr = t + (size_t)row * H;
    float s = 0.0f, s2 = 0.0f;
    for (int i = threadIdx.x; i < H; i += 256) { float v = xr[i]; s += v; s2 += v * v; }
    red[threadIdx.x] = s; __syncthreads();
    for (int o = 128; o; o >>= 1) { if (threadIdx.x < o) red[threadIdx.x] += red[threadIdx.x + o]; __syncthreads(); }
    const float mean = red[0] / H; __syncthreads();
    red[threadIdx.x] = s2; __syncthreads();
    for (int o = 128; o; o >>= 1) { if (threadIdx.x < o) red[threadIdx.x] += red[threadIdx.x + o]; __syncthreads(); }
    const float var = red[0] / H - mean * mean;
    const float rs = rsqrtf(var + 1e-5f);
    __syncthreads();
    float acc = 0.0f;
    for (int i = threadIdx.x; i < H; i += 256) {
        float v = (xr[i] - mean) * rs * g[i] + b[i];
        acc += v * w[i];
    }
    red[threadIdx.x] = acc; __syncthreads();
    for (int o = 128; o; o >>= 1) { if (threadIdx.x < o) red[threadIdx.x] += red[threadIdx.x + o]; __syncthreads(); }
    if (threadIdx.x == 0) outp[row] = red[0] + bias[0];
}

// ---------------------------------------------------------------------------
// Host orchestration
// ---------------------------------------------------------------------------
extern "C" void kernel_launch(void* const* d_in, const int* in_sizes, int n_in,
                              void* d_out, int out_size, void* d_ws, size_t ws_size,
                              hipStream_t stream)
{
    (void)in_sizes; (void)n_in; (void)out_size; (void)ws_size;

    constexpr int B = 32, Kk = 64, H = 2048, DE = 4096;
    constexpr int M1 = B * Kk;                 // 2048
    constexpr int NC = 200, NA = 10000, H4 = 4 * H;
    constexpr long long PREF_BSTRIDE = 258LL * 2048;

    const float* E_cate    = (const float*)d_in[0];
    const float* E_st      = (const float*)d_in[1];
    const float* E_hlong   = (const float*)d_in[2];
    const float* E_hshort  = (const float*)d_in[3];
    const float* E_next_st = (const float*)d_in[4];
    const float* hA        = (const float*)d_in[5];
    const float* hB        = (const float*)d_in[6];
    const float* proj_w1   = (const float*)d_in[7];
    const float* proj_b1   = (const float*)d_in[8];
    const float* proj_w2   = (const float*)d_in[9];
    const float* proj_b2   = (const float*)d_in[10];
    const float* poolA_w1  = (const float*)d_in[11];
    const float* poolA_b1  = (const float*)d_in[12];
    const float* poolA_w2  = (const float*)d_in[13];
    const float* poolA_b2  = (const float*)d_in[14];
    const float* poolB_w1  = (const float*)d_in[15];
    const float* poolB_b1  = (const float*)d_in[16];
    const float* poolB_w2  = (const float*)d_in[17];
    const float* poolB_b2  = (const float*)d_in[18];
    const float* hnc_w     = (const float*)d_in[19];
    const float* hnc_b     = (const float*)d_in[20];
    const float* happ_w    = (const float*)d_in[21];
    const float* happ_b    = (const float*)d_in[22];
    const float* ht_ln1_g  = (const float*)d_in[23];
    const float* ht_ln1_b  = (const float*)d_in[24];
    const float* ht_lin1_w = (const float*)d_in[25];
    const float* ht_lin1_b = (const float*)d_in[26];
    const float* ffn_norm_g= (const float*)d_in[27];
    const float* ffn_norm_b= (const float*)d_in[28];
    const float* ffn_w1    = (const float*)d_in[29];
    const float* ffn_b1    = (const float*)d_in[30];
    const float* ffn_w2    = (const float*)d_in[31];
    const float* ffn_b2    = (const float*)d_in[32];
    const float* ht_ln2_g  = (const float*)d_in[33];
    const float* ht_ln2_b  = (const float*)d_in[34];
    const float* ht_lin2_w = (const float*)d_in[35];
    const float* ht_lin2_b = (const float*)d_in[36];

    float* out       = (float*)d_out;
    float* o_app     = out;                       // [32,10000]
    float* o_traffic = out + 320000;              // [32,1]
    float* o_nc      = out + 320032;              // [32,200]
    float* o_prefA   = out + 326432;              // [32,258,2048]
    float* o_prefB   = o_prefA + 32LL * 258 * 2048;

    // ---- workspace bump allocator ----
    char* base = (char*)d_ws;
    size_t off = 0;
    auto alloc = [&](size_t bytes) -> void* {
        off = (off + 255) & ~(size_t)255;
        void* p = base + off;
        off += bytes;
        return p;
    };
    auto abf  = [&](size_t n) { return (bf16_t*)alloc(n * 2); };
    auto af32 = [&](size_t n) { return (float*)alloc(n * 4); };

    bf16_t* xa[4];
    for (int i = 0; i < 4; ++i) xa[i] = abf((size_t)M1 * H);
    bf16_t* xns = abf((size_t)B * H);
    bf16_t* xhA = abf((size_t)B * H);
    bf16_t* xhB = abf((size_t)B * H);
    bf16_t* w1T[6]; for (int i = 0; i < 6; ++i) w1T[i] = abf((size_t)DE * H);
    bf16_t* w2T[6]; for (int i = 0; i < 6; ++i) w2T[i] = abf((size_t)DE * DE);
    bf16_t* pAw1T = abf((size_t)H * H);
    bf16_t* pAw2T = abf((size_t)H * H);
    bf16_t* pBw1T = abf((size_t)H * H);
    bf16_t* pBw2T = abf((size_t)H * H);
    bf16_t* hncT  = abf((size_t)NC * H);
    bf16_t* happT = abf((size_t)NA * H);
    bf16_t* lin1T = abf((size_t)H * H);
    bf16_t* fw1T[2]; for (int i = 0; i < 2; ++i) fw1T[i] = abf((size_t)H4 * H);
    bf16_t* fw2T[2]; for (int i = 0; i < 2; ++i) fw2T[i] = abf((size_t)H * H4);
    bf16_t* mid       = abf((size_t)M1 * DE);
    bf16_t* mid32     = abf((size_t)B * DE);
    bf16_t* poolmid   = abf((size_t)B * H);
    bf16_t* pooledAbf = abf((size_t)B * H);
    bf16_t* pooledBbf = abf((size_t)B * H);
    bf16_t* u_bf      = abf((size_t)B * H);
    bf16_t* ffnmid    = abf((size_t)B * H4);
    float*  pooledAf  = af32((size_t)B * H);
    float*  pooledBf  = af32((size_t)B * H);
    float*  tbuf      = af32((size_t)B * H);

    // ---- launch helpers ----
    auto tconv = [&](const float* in, bf16_t* o, int R, int C) {
        dim3 g((C + 31) / 32, (R + 31) / 32);
        transpose_cvt<<<g, 256, 0, stream>>>(in, o, R, C);
    };
    auto cvt = [&](const float* in, bf16_t* o, size_t n) {
        cvt_f32_bf16<<<(int)((n + 255) / 256), 256, 0, stream>>>(in, o, (int)n);
    };
    auto gemm = [&](int ep, const bf16_t* A, const bf16_t* Bt, const float* bias,
                    const float* resid, void* C, int M, int N, int K,
                    int rpb, long long bstr) {
        dim3 g((N + 127) / 128, (M + 127) / 128);
        const bool full = (M % 128 == 0) && (N % 128 == 0);
        if (full) {
            switch (ep) {
            case 0: gemm_bf16_wmma_async<0><<<g, 256, 0, stream>>>(A, Bt, bias, resid, C, M, N, K, rpb, bstr); break;
            case 1: gemm_bf16_wmma_async<1><<<g, 256, 0, stream>>>(A, Bt, bias, resid, C, M, N, K, rpb, bstr); break;
            default: gemm_bf16_wmma_async<2><<<g, 256, 0, stream>>>(A, Bt, bias, resid, C, M, N, K, rpb, bstr); break;
            }
        } else {
            switch (ep) {
            case 0: gemm_bf16_wmma_guard<0><<<g, 256, 0, stream>>>(A, Bt, bias, resid, C, M, N, K, rpb, bstr); break;
            case 1: gemm_bf16_wmma_guard<1><<<g, 256, 0, stream>>>(A, Bt, bias, resid, C, M, N, K, rpb, bstr); break;
            default: gemm_bf16_wmma_guard<2><<<g, 256, 0, stream>>>(A, Bt, bias, resid, C, M, N, K, rpb, bstr); break;
            }
        }
    };
    auto ln = [&](const float* x, const float* g, const float* b, bf16_t* y, int rows) {
        layernorm_to_bf16<<<rows, 256, 0, stream>>>(x, g, b, y, H);
    };

    // ---- 1) weight transpose + downconvert (once per launch) ----
    for (int i = 0; i < 6; ++i) {
        tconv(proj_w1 + (size_t)i * H * DE, w1T[i], H, DE);
        tconv(proj_w2 + (size_t)i * DE * DE, w2T[i], DE, DE);
    }
    tconv(poolA_w1, pAw1T, H, H);  tconv(poolA_w2, pAw2T, H, H);
    tconv(poolB_w1, pBw1T, H, H);  tconv(poolB_w2, pBw2T, H, H);
    tconv(hnc_w, hncT, H, NC);
    tconv(happ_w, happT, H, NA);
    tconv(ht_lin1_w, lin1T, H, H);
    for (int i = 0; i < 2; ++i) {
        tconv(ffn_w1 + (size_t)i * H * H4, fw1T[i], H, H4);
        tconv(ffn_w2 + (size_t)i * H4 * H, fw2T[i], H4, H);
    }

    // ---- 2) activation downconvert ----
    const float* Xin[4] = {E_cate, E_st, E_hlong, E_hshort};
    for (int i = 0; i < 4; ++i) cvt(Xin[i], xa[i], (size_t)M1 * H);
    cvt(E_next_st, xns, (size_t)B * H);
    cvt(hA, xhA, (size_t)B * H);
    cvt(hB, xhB, (size_t)B * H);

    // ---- 3) four big projectors (M=2048), writing straight into pref layout ----
    float* pdst[4] = {o_prefA, o_prefA + 128 * 2048, o_prefB + 2 * 2048, o_prefB + 130 * 2048};
    for (int i = 0; i < 4; ++i) {
        gemm(2, xa[i], w1T[i], proj_b1 + (size_t)i * DE, nullptr, mid, M1, DE, H, M1, 0);
        gemm(0, mid, w2T[i], proj_b2 + (size_t)i * DE, nullptr, pdst[i], M1, DE, DE, 64, PREF_BSTRIDE);
    }

    // ---- 4) P_nextst (slot 4, M=32) -> prefA rows 256..257 per batch ----
    gemm(2, xns, w1T[4], proj_b1 + 4 * (size_t)DE, nullptr, mid32, B, DE, H, B, 0);
    gemm(0, mid32, w2T[4], proj_b2 + 4 * (size_t)DE, nullptr, o_prefA + 256 * 2048, B, DE, DE, 1, PREF_BSTRIDE);

    // ---- 5) P_nextcate (slot 5, input hA) -> prefB rows 0..1 per batch ----
    gemm(2, xhA, w1T[5], proj_b1 + 5 * (size_t)DE, nullptr, mid32, B, DE, H, B, 0);
    gemm(0, mid32, w2T[5], proj_b2 + 5 * (size_t)DE, nullptr, o_prefB, B, DE, DE, 1, PREF_BSTRIDE);

    // ---- 6) pool A -> logits_next_cate ----
    gemm(2, xhA, pAw1T, poolA_b1, nullptr, poolmid, B, H, H, B, 0);
    gemm(0, poolmid, pAw2T, poolA_b2, nullptr, pooledAf, B, H, H, B, 0);
    cvt(pooledAf, pooledAbf, (size_t)B * H);
    gemm(0, pooledAbf, hncT, hnc_b, nullptr, o_nc, B, NC, H, B, 0);

    // ---- 7) pool B -> logits_app ----
    gemm(2, xhB, pBw1T, poolB_b1, nullptr, poolmid, B, H, H, B, 0);
    gemm(0, poolmid, pBw2T, poolB_b2, nullptr, pooledBf, B, H, H, B, 0);
    cvt(pooledBf, pooledBbf, (size_t)B * H);
    gemm(0, pooledBbf, happT, happ_b, nullptr, o_app, B, NA, H, B, 0);

    // ---- 8) head_traffic: LN -> lin1 -> 2x FFN block -> LN -> dot ----
    ln(pooledBf, ht_ln1_g, ht_ln1_b, u_bf, B);
    gemm(0, u_bf, lin1T, ht_lin1_b, nullptr, tbuf, B, H, H, B, 0);
    for (int i = 0; i < 2; ++i) {
        ln(tbuf, ffn_norm_g + (size_t)i * H, ffn_norm_b + (size_t)i * H, u_bf, B);
        gemm(2, u_bf, fw1T[i], ffn_b1 + (size_t)i * H4, nullptr, ffnmid, B, H4, H, B, 0);
        gemm(1, ffnmid, fw2T[i], ffn_b2 + (size_t)i * H, tbuf, tbuf, B, H, H4, B, 0);
    }
    traffic_head<<<B, 256, 0, stream>>>(tbuf, ht_ln2_g, ht_ln2_b, ht_lin2_w, ht_lin2_b,
                                        o_traffic, H);
}
